// S5Layer_76209899700679
// MI455X (gfx1250) — matrix operand order repaired and checked by reading, not compiled, not run on hardware
//
#include <hip/hip_runtime.h>
#include <stdint.h>

// ---------------- problem constants ----------------
#define T_LEN  4096
#define B_SZ   16
#define P_SZ   512
#define H_SZ   512
#define LC     64                   // scan chunk length
#define NCHUNK (T_LEN / LC)         // 64 chunks
#define NPT    256                  // p-tile per workgroup (2 tiles cover P)
#define SB_STRIDE 264               // 256 + 8 bf16 pad -> 528B rows (TDM pads)
#define SA_STRIDE 528               // 512 + 16 bf16 pad -> 1056B rows

typedef __attribute__((ext_vector_type(16))) __bf16 v16bf;
typedef __attribute__((ext_vector_type(8)))  __bf16 v8bf;
typedef __attribute__((ext_vector_type(8)))  float  v8f;
typedef unsigned int u32x4 __attribute__((ext_vector_type(4)));
typedef int          i32x8 __attribute__((ext_vector_type(8)));
typedef int          i32x4 __attribute__((ext_vector_type(4)));

// round-to-nearest-even f32 -> bf16 (stored as ushort)
__device__ __forceinline__ unsigned short f2bf(float x) {
  unsigned u = __float_as_uint(x);
  u += 0x7FFFu + ((u >> 16) & 1u);
  return (unsigned short)(u >> 16);
}

// ---------------- kernel 0: discretization + power table + b_bar^T (bf16) ---
__global__ void prep_kernel(const float* __restrict__ log_neg_a,
                            const float* __restrict__ bmat,     // [P,H]
                            const float* __restrict__ log_dt,
                            float* __restrict__ a_bar,          // [P]
                            float* __restrict__ abar_pow,       // [LC+1][P]
                            unsigned short* __restrict__ bT) {  // [H][P] bf16
  int p = blockIdx.x * blockDim.x + threadIdx.x;
  if (p >= P_SZ) return;
  float dt   = __expf(log_dt[p]);
  float na   = __expf(log_neg_a[p]);   // -a
  float a    = -na;
  float abar = __expf(dt * a);
  a_bar[p] = abar;
  float pw = 1.f;
  for (int j = 0; j <= LC; ++j) { abar_pow[j * P_SZ + p] = pw; pw *= abar; }
  float scale = ((abar - 1.f) / a) * dt;
  for (int h = 0; h < H_SZ; ++h)
    bT[h * P_SZ + p] = f2bf(scale * bmat[p * H_SZ + h]);
}

// ---------------- kernel 1: fused GEMM(WMMA-bf16) + chunk-local scan --------
// grid = (P/NPT = 2, NCHUNK = 64); block = 256 threads (8 wave32)
// wave w owns p-columns [p0+32w, p0+32w+31] as TWO 16x16 f32 accumulators.
// One 16x16 WMMA D-tile == one timestep t (16 batches x 16 p), so the S5
// recurrence z = a_bar*z + bu_t runs entirely inside the accumulators.
// A-panel staging is software-pipelined through a double-buffered LDS slab.
__global__ __launch_bounds__(256) void fused_kernel(
    const float* __restrict__ in,           // [T,B,H] f32
    const float* __restrict__ a_bar,        // [P]
    const unsigned short* __restrict__ bT,  // [H][P] bf16
    float* __restrict__ out,                // [T,B,P] f32
    float* __restrict__ carry) {            // [NCHUNK,B,P] f32
  __shared__ unsigned short sB[H_SZ * SB_STRIDE];      // 270,336 B (512 x 256)
  __shared__ unsigned short sA[2][B_SZ * SA_STRIDE];   //  33,792 B (2x 16x512)

  const int tid  = threadIdx.x;
  const int wave = tid >> 5;
  const int lane = tid & 31;
  const int col  = lane & 15;      // N within 16x16 tile
  const int hi   = lane >> 4;      // half-wave select (M rows 0-7 vs 8-15)
  const int p0   = blockIdx.x * NPT;
  const int t0   = blockIdx.y * LC;

  // ---- TDM: DMA whole B tile (512 rows x 256 bf16) global->LDS, one op.
  // pad_enable inserts 16B after every 512B row => LDS row stride 528B,
  // matching SB_STRIDE and de-conflicting LDS banks for the WMMA-B reads.
  if (wave == 0) {
    unsigned lds_off = (unsigned)(uintptr_t)(void*)&sB[0];
    uint64_t ga = (uint64_t)(uintptr_t)(bT + p0);       // tile start address
    u32x4 g0; i32x8 g1;
    i32x4 gz4 = {0, 0, 0, 0};
    i32x8 gz8 = {0, 0, 0, 0, 0, 0, 0, 0};
    g0[0] = 1u;                                          // count = 1
    g0[1] = lds_off;                                     // lds_addr
    g0[2] = (unsigned)ga;                                // global_addr lo
    g0[3] = (unsigned)((ga >> 32) & 0x1FFFFFFu) | (2u << 30);  // hi | type=2
    g1[0] = (int)((1u << 16)        // data_size = 2 bytes
                | (1u << 20)        // pad_enable
                | (6u << 22)        // pad_interval: every 128 DW (512B)
                | (3u << 25));      // pad_amount : 4 DW (16B)
    g1[1] = (int)((unsigned)H_SZ << 16);   // tensor_dim0 = 512 (row length)
    g1[2] = (int)((unsigned)H_SZ << 16);   // dim0 hi=0 | tensor_dim1 = 512
    g1[3] = (int)((unsigned)NPT << 16);    // dim1 hi=0 | tile_dim0 = 256
    g1[4] = (int)512u;                     // tile_dim1 = 512, tile_dim2 = 0
    g1[5] = (int)P_SZ;                     // tensor_dim0_stride = 512 elems
    g1[6] = 0;
    g1[7] = 0;
    __builtin_amdgcn_tensor_load_to_lds(g0, g1, gz4, gz4, gz8, 0);
    __builtin_amdgcn_s_wait_tensorcnt(0);
  }

  const float am0 = a_bar[p0 + wave * 32 + col];        // per-lane a_bar
  const float am1 = a_bar[p0 + wave * 32 + 16 + col];
  v8f acc0 = {0.f, 0.f, 0.f, 0.f, 0.f, 0.f, 0.f, 0.f};
  v8f acc1 = {0.f, 0.f, 0.f, 0.f, 0.f, 0.f, 0.f, 0.f};

  // A staging assignment: thread stages 32 consecutive h of one batch row.
  const int arow = tid >> 4;          // batch row this thread stages
  const int acb  = (tid & 15) * 32;   // 32 h-values per thread
  const float* abase = in + ((size_t)t0 * B_SZ + arow) * H_SZ + acb;

  // software pipeline: preload A(t=0) into registers
  float4 reg[8];
#pragma unroll
  for (int i = 0; i < 8; ++i) reg[i] = *(const float4*)(abase + 4 * i);

  for (int t = 0; t < LC; ++t) {
    // drain pipeline stage: convert A(t) regs -> bf16 in sA[t&1]
    unsigned short* dst = &sA[t & 1][arow * SA_STRIDE + acb];
#pragma unroll
    for (int i = 0; i < 8; ++i) {
      float4 f = reg[i];
      dst[4 * i + 0] = f2bf(f.x);
      dst[4 * i + 1] = f2bf(f.y);
      dst[4 * i + 2] = f2bf(f.z);
      dst[4 * i + 3] = f2bf(f.w);
    }
    __syncthreads();   // sA[t&1] visible; all readers of sA[(t+1)&1] done

    // issue global loads for A(t+1); latency hides behind the WMMA chain
    if (t + 1 < LC) {
      const float* nsrc = abase + (size_t)(t + 1) * B_SZ * H_SZ;
#pragma unroll
      for (int i = 0; i < 8; ++i) reg[i] = *(const float4*)(nsrc + 4 * i);
    }

    // scan step: z <- a_bar * z  (then WMMAs add bu_t on top)
#pragma unroll
    for (int i = 0; i < 8; ++i) { acc0[i] *= am0; acc1[i] *= am1; }

    const unsigned short* sAt = &sA[t & 1][0];
#pragma unroll
    for (int k0 = 0; k0 < H_SZ; k0 += 32) {
      // A frag: lane<16 holds row M=lane, K {k0..k0+7, k0+16..k0+23};
      //         lane>=16 holds M=lane-16, K {k0+8..15, k0+24..31}
      union { v16bf v; v8bf h[2]; } af, bw0, bw1;
      af.h[0] = *(const v8bf*)&sAt[(lane & 15) * SA_STRIDE + k0 + 8 * hi];
      af.h[1] = *(const v8bf*)&sAt[(lane & 15) * SA_STRIDE + k0 + 16 + 8 * hi];
      // B frags: lane holds column N=col, K {k0+8hi..+7, k0+16+8hi..+7}
      const unsigned short* brow = &sB[(k0 + 8 * hi) * SB_STRIDE + wave * 32 + col];
#pragma unroll
      for (int s = 0; s < 16; ++s) {
        const int kk = (s & 7) + ((s >> 3) << 4);
        bw0.v[s] = __builtin_bit_cast(__bf16, brow[kk * SB_STRIDE]);
        bw1.v[s] = __builtin_bit_cast(__bf16, brow[kk * SB_STRIDE + 16]);
      }
      acc0 = __builtin_amdgcn_wmma_f32_16x16x32_bf16(
          false, af.v, false, bw0.v, (short)0, acc0, false, false);
      acc1 = __builtin_amdgcn_wmma_f32_16x16x32_bf16(
          false, af.v, false, bw1.v, (short)0, acc1, false, false);
    }

    // store z_t: VGPR r = batch (hi*8 + r), columns p0+32w+col and +16
    float* orow = out + ((size_t)(t0 + t) * B_SZ + hi * 8) * P_SZ
                + p0 + wave * 32 + col;
#pragma unroll
    for (int r = 0; r < 8; ++r) {
      orow[(size_t)r * P_SZ]      = acc0[r];
      orow[(size_t)r * P_SZ + 16] = acc1[r];
    }
  }

  // chunk-local final state -> carry
  float* crow = carry + ((size_t)blockIdx.y * B_SZ + hi * 8) * P_SZ
              + p0 + wave * 32 + col;
#pragma unroll
  for (int r = 0; r < 8; ++r) {
    crow[(size_t)r * P_SZ]      = acc0[r];
    crow[(size_t)r * P_SZ + 16] = acc1[r];
  }
}

// ---------------- kernel 2: sequential combine over 64 chunk carries -------
__global__ void carry_scan_kernel(const float* __restrict__ carry,
                                  const float* __restrict__ abar_pow,
                                  float* __restrict__ carry_in) {
  int idx = blockIdx.x * blockDim.x + threadIdx.x;  // 0..B*P-1
  int p = idx & (P_SZ - 1);
  int b = idx >> 9;
  float apL  = abar_pow[LC * P_SZ + p];             // a_bar^LC
  float zend = 0.f;
  for (int c = 0; c < NCHUNK; ++c) {
    size_t o = ((size_t)c * B_SZ + b) * P_SZ + p;
    carry_in[o] = zend;                              // state entering chunk c
    zend = carry[o] + apL * zend;
  }
}

// ---------------- kernel 3: parallel fixup  z += a_bar^(j+1) * carry_in ----
__global__ __launch_bounds__(256) void fixup_kernel(
    float* __restrict__ out,
    const float* __restrict__ abar_pow,
    const float* __restrict__ carry_in) {
  size_t idx = (size_t)blockIdx.x * blockDim.x + threadIdx.x;
  int    p  = (int)(idx & (P_SZ - 1));
  size_t bp = idx >> 9;
  int    b  = (int)(bp & (B_SZ - 1));
  size_t t  = bp >> 4;
  int    c  = (int)(t >> 6);          // chunk (LC = 64)
  int    j  = (int)(t & (LC - 1));    // offset within chunk
  out[idx] += abar_pow[(j + 1) * P_SZ + p]
            * carry_in[((size_t)c * B_SZ + b) * P_SZ + p];
}

// ---------------- launcher --------------------------------------------------
extern "C" void kernel_launch(void* const* d_in, const int* in_sizes, int n_in,
                              void* d_out, int out_size, void* d_ws, size_t ws_size,
                              hipStream_t stream) {
  const float* inputs    = (const float*)d_in[0];   // [T,B,H]
  const float* log_neg_a = (const float*)d_in[1];   // [P]
  const float* bmat      = (const float*)d_in[2];   // [P,H]
  const float* log_dt    = (const float*)d_in[3];   // [P]
  float* out = (float*)d_out;
  char*  ws  = (char*)d_ws;

  // workspace layout (~4.6 MB total)
  float*          a_bar    = (float*)(ws);                               //   2 KB
  float*          abar_pow = (float*)(ws + 2048);                        // 130 KB
  unsigned short* bT       = (unsigned short*)(ws + 135168);             // 512 KB
  float*          carry    = (float*)(ws + 659456);                      //   2 MB
  float*          carry_in = (float*)(ws + 2756608);                     //   2 MB

  prep_kernel<<<2, 256, 0, stream>>>(log_neg_a, bmat, log_dt,
                                     a_bar, abar_pow, bT);
  fused_kernel<<<dim3(P_SZ / NPT, NCHUNK), 256, 0, stream>>>(
      inputs, a_bar, bT, out, carry);
  carry_scan_kernel<<<(B_SZ * P_SZ) / 256, 256, 0, stream>>>(
      carry, abar_pow, carry_in);
  fixup_kernel<<<(unsigned)((size_t)T_LEN * B_SZ * P_SZ / 256), 256, 0, stream>>>(
      out, abar_pow, carry_in);
}
